// SkipMamba_14353780703877
// MI455X (gfx1250) — matrix operand order repaired and checked by reading, not compiled, not run on hardware
//
#include <hip/hip_runtime.h>
#include <hip/hip_bf16.h>
#include <math.h>

// ---------------------------------------------------------------------------
// BiMamba skip block for MI455X (gfx1250, wave32).
// GEMMs use v_wmma_f32_16x16x32_bf16; selective scan is one wave per (b,d).
// ---------------------------------------------------------------------------

typedef __bf16 bf16_t;
typedef __attribute__((ext_vector_type(16))) __bf16 v16bf;
typedef __attribute__((ext_vector_type(8)))  __bf16 v8bf;
typedef __attribute__((ext_vector_type(8)))  float  v8f;

#define BB   8
#define CIN  12
#define LL   8192
#define HH   32
#define DIN  64      // d_inner
#define NS   16      // d_state
#define MTOT (BB * LL)   // 65536 rows

// ---------------------------------------------------------------------------
// Prep: bf16 transposed weights, A = -exp(A_log), weight-normed w1/w2
// ---------------------------------------------------------------------------
__global__ void prep_kernel(const float* __restrict__ inprj_f, const float* __restrict__ inprj_b,
                            const float* __restrict__ xprj_f,  const float* __restrict__ xprj_b,
                            const float* __restrict__ oprj_f,  const float* __restrict__ oprj_b,
                            const float* __restrict__ alog_f,  const float* __restrict__ alog_b,
                            const float* __restrict__ c1v, const float* __restrict__ c1g,
                            const float* __restrict__ c2v, const float* __restrict__ c2g,
                            bf16_t* __restrict__ WtIn,   // 2 x (32 x 128)
                            bf16_t* __restrict__ WtX,    // 2 x (64 x 48)  (34 cols valid, rest 0)
                            bf16_t* __restrict__ WtOut,  // 2 x (64 x 32)
                            float*  __restrict__ Aneg,   // 2 x (64 x 16)
                            float*  __restrict__ w1n,    // 32 x 64
                            float*  __restrict__ w2n)    // 4 x 32
{
    int tid = threadIdx.x;
    for (int i = tid; i < 32 * 128; i += 256) {
        int k = i / 128, n = i % 128;
        WtIn[i]            = (bf16_t)inprj_f[n * 32 + k];
        WtIn[32 * 128 + i] = (bf16_t)inprj_b[n * 32 + k];
    }
    for (int i = tid; i < 64 * 48; i += 256) {
        int k = i / 48, n = i % 48;
        float vf = (n < 34) ? xprj_f[n * 64 + k] : 0.f;
        float vb = (n < 34) ? xprj_b[n * 64 + k] : 0.f;
        WtX[i]           = (bf16_t)vf;
        WtX[64 * 48 + i] = (bf16_t)vb;
    }
    for (int i = tid; i < 64 * 32; i += 256) {
        int k = i / 32, n = i % 32;
        WtOut[i]           = (bf16_t)oprj_f[n * 64 + k];
        WtOut[64 * 32 + i] = (bf16_t)oprj_b[n * 64 + k];
    }
    for (int i = tid; i < DIN * NS; i += 256) {
        Aneg[i]            = -__expf(alog_f[i]);
        Aneg[DIN * NS + i] = -__expf(alog_b[i]);
    }
    if (tid < 32) {  // weight_norm dim=0 over (in,kw)
        float s = 0.f;
        for (int c = 0; c < 64; ++c) { float v = c1v[tid * 64 + c]; s += v * v; }
        float sc = c1g[tid] * rsqrtf(s);
        for (int c = 0; c < 64; ++c) w1n[tid * 64 + c] = c1v[tid * 64 + c] * sc;
    }
    if (tid < 4) {
        float s = 0.f;
        for (int c = 0; c < 32; ++c) { float v = c2v[tid * 32 + c]; s += v * v; }
        float sc = c2g[tid] * rsqrtf(s);
        for (int c = 0; c < 32; ++c) w2n[tid * 32 + c] = c2v[tid * 32 + c] * sc;
    }
}

// ---------------------------------------------------------------------------
// Input projection: hs[b,l,h] = proj_w @ x + proj_b  (bf16 out, WMMA A feed)
// ---------------------------------------------------------------------------
__global__ void proj_kernel(const float* __restrict__ x, const float* __restrict__ pw,
                            const float* __restrict__ pb, bf16_t* __restrict__ hs)
{
    int idx = blockIdx.x * blockDim.x + threadIdx.x;   // over MTOT*32
    if (idx >= MTOT * HH) return;
    int h  = idx & 31;
    int ml = idx >> 5;
    int b = ml / LL, l = ml % LL;
    float s = pb[h];
#pragma unroll
    for (int c = 0; c < CIN; ++c)
        s += pw[h * CIN + c] * x[((size_t)(b * CIN + c)) * LL + l];
    hs[(size_t)ml * HH + h] = (bf16_t)s;
}

// ---------------------------------------------------------------------------
// Generic tall-skinny GEMM: C[m,n] = sum_k A[m,k] * Wt[k,n]   (A bf16, Wt bf16)
// One wave -> one 16x16 C tile via v_wmma_f32_16x16x32_bf16; KSZ in {32,64}.
// reverse!=0 re-indexes A rows l -> L-1-l within each batch (backward mamba).
// ---------------------------------------------------------------------------
template <int KSZ, int STOREBF>
__global__ void wmma_gemm_kernel(const bf16_t* __restrict__ A,
                                 const bf16_t* __restrict__ Wt,
                                 float* __restrict__ Cf, bf16_t* __restrict__ Cb,
                                 int Npad, int ldc, int colOff, int reverse)
{
    int lane  = threadIdx.x & 31;
    int wave  = threadIdx.x >> 5;
    int tileM = blockIdx.x * 4 + wave;
    int tileN = blockIdx.y;
    int half  = lane >> 4;
    int r16   = lane & 15;

    // A fragment: lane (half,r16) holds row r16 of the tile, K split per ISA layout
    int mrow = tileM * 16 + r16;
    int b = mrow / LL, l = mrow % LL;
    if (reverse) l = LL - 1 - l;
    size_t arow = ((size_t)b * LL + l) * (size_t)KSZ;

    v8f acc = {};
#pragma unroll
    for (int ks = 0; ks < KSZ; ks += 32) {
        int ka0 = ks + half * 8;        // K 0..7  (half 0) / 8..15  (half 1)
        int ka1 = ks + 16 + half * 8;   // K 16..23         / 24..31
        v8bf a_lo = *(const v8bf*)(A + arow + ka0);
        v8bf a_hi = *(const v8bf*)(A + arow + ka1);
        v16bf afrag;
#pragma unroll
        for (int i = 0; i < 8; ++i) { afrag[i] = a_lo[i]; afrag[8 + i] = a_hi[i]; }
        // B fragment: lane holds column N = r16, K in the same split as A (ka0.., ka1..)
        v16bf bfrag;
        const bf16_t* bcol = Wt + tileN * 16 + r16;
#pragma unroll
        for (int i = 0; i < 8; ++i) {
            bfrag[i]     = bcol[(size_t)(ka0 + i) * Npad];
            bfrag[8 + i] = bcol[(size_t)(ka1 + i) * Npad];
        }
        acc = __builtin_amdgcn_wmma_f32_16x16x32_bf16(false, afrag, false, bfrag,
                                                      (short)0, acc, false, false);
    }
    // C/D layout: VGPR r -> row r (lanes 0-15) / r+8 (lanes 16-31), N = lane&15
    int ncol  = tileN * 16 + r16 + colOff;
    int mbase = tileM * 16 + half * 8;
#pragma unroll
    for (int r = 0; r < 8; ++r) {
        size_t off = (size_t)(mbase + r) * ldc + ncol;
        if (STOREBF) Cb[off] = (bf16_t)acc[r];
        else         Cf[off] = acc[r];
    }
}

// ---------------------------------------------------------------------------
// Depthwise causal conv (K=4) + bias + SiLU over xs = xz[:, :, 0:64]
// ---------------------------------------------------------------------------
__global__ void conv_silu_kernel(const bf16_t* __restrict__ xz,   // (M,128)
                                 const float* __restrict__ cw, const float* __restrict__ cb,
                                 bf16_t* __restrict__ xc)          // (M,64)
{
    int idx = blockIdx.x * blockDim.x + threadIdx.x;   // over MTOT*64
    if (idx >= MTOT * DIN) return;
    int d  = idx & 63;
    int ml = idx >> 6;
    int b = ml / LL, l = ml % LL;
    float s = cb[d];
#pragma unroll
    for (int j = 0; j < 4; ++j) {
        int li = l - 3 + j;
        if (li >= 0)
            s += cw[d * 4 + j] * (float)xz[((size_t)(b * LL + li)) * 128 + d];
    }
    float sl = s / (1.f + __expf(-s));   // silu
    xc[(size_t)ml * DIN + d] = (bf16_t)sl;
}

// ---------------------------------------------------------------------------
// Selective scan: one wave per (b,d); 16 lanes = n-states, f32 state.
// Fuses dt_proj+softplus, recurrence, C-dot (shfl butterfly), +u*D, silu(z) gate.
// ---------------------------------------------------------------------------
__global__ void scan_kernel(const float* __restrict__ xdbl,   // (M,48): dt0,dt1,B[16],C[16],pad
                            const bf16_t* __restrict__ xc,    // (M,64) u
                            const bf16_t* __restrict__ xz,    // (M,128) z in cols 64..127
                            const float* __restrict__ Aneg,   // (64,16)
                            const float* __restrict__ dtw,    // (64,2)
                            const float* __restrict__ dtb,    // (64)
                            const float* __restrict__ Dp,     // (64)
                            bf16_t* __restrict__ yg)          // (M,64)
{
    int lane = threadIdx.x & 31;
    int wid  = (blockIdx.x * blockDim.x + threadIdx.x) >> 5;
    int b = wid / DIN, d = wid % DIN;
    int n = lane & 15;

    float Adn = Aneg[d * NS + n];
    float w0 = dtw[d * 2 + 0], w1 = dtw[d * 2 + 1];
    float bb = dtb[d], Dd = Dp[d];
    float h = 0.f;
    size_t row0 = (size_t)b * LL;

    for (int l = 0; l < LL; ++l) {
        size_t m = row0 + l;
        const float* xr = xdbl + m * 48;
        if (l + 8 < LL) __builtin_prefetch(xdbl + (m + 8) * 48, 0, 1);  // global_prefetch_b8
        float dtv = xr[0] * w0 + xr[1] * w1 + bb;
        float dt  = (dtv > 20.f) ? dtv : log1pf(__expf(dtv));           // softplus
        float Bn = xr[2 + n];
        float Cn = xr[18 + n];
        float u  = (float)xc[m * DIN + d];
        float z  = (float)xz[m * 128 + 64 + d];
        h = __expf(dt * Adn) * h + (dt * u) * Bn;
        float p = h * Cn;
        p += __shfl_xor(p, 8, 32);
        p += __shfl_xor(p, 4, 32);
        p += __shfl_xor(p, 2, 32);
        p += __shfl_xor(p, 1, 32);
        float y = p + u * Dd;
        float g = z / (1.f + __expf(-z));
        if (lane == 0) yg[m * DIN + d] = (bf16_t)(y * g);
    }
}

// ---------------------------------------------------------------------------
// Final head: one wave per (b,l); lane = h. z1 = tanh((w1n@y + ident)/sqrt2),
// out = w2n@z1 + b2 (4 shuffle reductions).
// ---------------------------------------------------------------------------
__global__ void final_kernel(const float* __restrict__ ycat,   // (M,64)
                             const float* __restrict__ x,
                             const float* __restrict__ w1n, const float* __restrict__ b1,
                             const float* __restrict__ iw,  const float* __restrict__ ib,
                             const float* __restrict__ w2n, const float* __restrict__ b2,
                             float* __restrict__ out)           // (B,4,L)
{
    int lane = threadIdx.x & 31;
    int ml   = (blockIdx.x * blockDim.x + threadIdx.x) >> 5;
    int b = ml / LL, l = ml % LL;
    const float* yr = ycat + (size_t)ml * 64;
    float s = b1[lane] + ib[lane];
    for (int c = 0; c < 64; ++c) s += w1n[lane * 64 + c] * yr[c];
#pragma unroll
    for (int c = 0; c < CIN; ++c)
        s += iw[lane * CIN + c] * x[((size_t)(b * CIN + c)) * LL + l];
    float z1 = tanhf(s * (1.0f / 1.41421f));
#pragma unroll
    for (int o = 0; o < 4; ++o) {
        float p = w2n[o * 32 + lane] * z1;
        p += __shfl_xor(p, 16, 32);
        p += __shfl_xor(p, 8, 32);
        p += __shfl_xor(p, 4, 32);
        p += __shfl_xor(p, 2, 32);
        p += __shfl_xor(p, 1, 32);
        if (lane == 0) out[((size_t)(b * 4 + o)) * LL + l] = p + b2[o];
    }
}

// ---------------------------------------------------------------------------
// Host orchestration
// ---------------------------------------------------------------------------
extern "C" void kernel_launch(void* const* d_in, const int* in_sizes, int n_in,
                              void* d_out, int out_size, void* d_ws, size_t ws_size,
                              hipStream_t stream) {
    if (n_in < 29 || !d_out || !d_ws) return;
    (void)in_sizes; (void)out_size; (void)ws_size;

    // setup_inputs() insertion order (nested dicts depth-first)
    const float* x      = (const float*)d_in[0];
    const float* proj_w = (const float*)d_in[1];
    const float* proj_b = (const float*)d_in[2];
    const float *m_inprj[2], *m_convw[2], *m_convb[2], *m_xprj[2], *m_dtw[2],
                *m_dtb[2], *m_alog[2], *m_D[2], *m_oprj[2];
    for (int dir = 0; dir < 2; ++dir) {     // dir 0 = mamba_f, dir 1 = mamba_b
        int base = 3 + dir * 9;
        m_inprj[dir] = (const float*)d_in[base + 0];
        m_convw[dir] = (const float*)d_in[base + 1];
        m_convb[dir] = (const float*)d_in[base + 2];
        m_xprj[dir]  = (const float*)d_in[base + 3];
        m_dtw[dir]   = (const float*)d_in[base + 4];
        m_dtb[dir]   = (const float*)d_in[base + 5];
        m_alog[dir]  = (const float*)d_in[base + 6];
        m_D[dir]     = (const float*)d_in[base + 7];
        m_oprj[dir]  = (const float*)d_in[base + 8];
    }
    const float* c1v = (const float*)d_in[21];
    const float* c1g = (const float*)d_in[22];
    const float* c1b = (const float*)d_in[23];
    const float* c2v = (const float*)d_in[24];
    const float* c2g = (const float*)d_in[25];
    const float* c2b = (const float*)d_in[26];
    const float* idw = (const float*)d_in[27];
    const float* idb = (const float*)d_in[28];

    // workspace layout
    char* ws = (char*)d_ws;
    size_t off = 0;
    auto take = [&](size_t bytes) -> void* {
        void* p = ws + off;
        off = (off + bytes + 255) & ~(size_t)255;
        return p;
    };
    bf16_t* hs_bf = (bf16_t*)take((size_t)MTOT * HH * 2);
    bf16_t* xz_bf[2];  bf16_t* xc_bf[2];  float* xdbl[2];  bf16_t* yg_bf[2];
    for (int dir = 0; dir < 2; ++dir) {
        xz_bf[dir] = (bf16_t*)take((size_t)MTOT * 128 * 2);
        xc_bf[dir] = (bf16_t*)take((size_t)MTOT * DIN * 2);
        xdbl[dir]  = (float*) take((size_t)MTOT * 48  * 4);
        yg_bf[dir] = (bf16_t*)take((size_t)MTOT * DIN * 2);
    }
    float*  ycat  = (float*) take((size_t)MTOT * 64 * 4);
    bf16_t* WtIn  = (bf16_t*)take(2 * 32 * 128 * 2);
    bf16_t* WtX   = (bf16_t*)take(2 * 64 * 48  * 2);
    bf16_t* WtOut = (bf16_t*)take(2 * 64 * 32  * 2);
    float*  Aneg  = (float*) take(2 * DIN * NS * 4);
    float*  w1n   = (float*) take(32 * 64 * 4);
    float*  w2n   = (float*) take(4 * 32 * 4);

    prep_kernel<<<1, 256, 0, stream>>>(m_inprj[0], m_inprj[1], m_xprj[0], m_xprj[1],
                                       m_oprj[0], m_oprj[1], m_alog[0], m_alog[1],
                                       c1v, c1g, c2v, c2g,
                                       WtIn, WtX, WtOut, Aneg, w1n, w2n);

    proj_kernel<<<(MTOT * HH) / 256, 256, 0, stream>>>(x, proj_w, proj_b, hs_bf);

    const int MT4 = MTOT / 16 / 4;   // 1024 blocks of 4 M-tiles
    for (int dir = 0; dir < 2; ++dir) {
        // in_proj: (M,32) @ (32,128) -> bf16 xz ; backward reads reversed rows
        wmma_gemm_kernel<32, 1><<<dim3(MT4, 8), 128, 0, stream>>>(
            hs_bf, WtIn + dir * 32 * 128, nullptr, xz_bf[dir], 128, 128, 0, dir);
        // depthwise conv + silu -> xc
        conv_silu_kernel<<<(MTOT * DIN) / 256, 256, 0, stream>>>(
            xz_bf[dir], m_convw[dir], m_convb[dir], xc_bf[dir]);
        // x_proj: (M,64) @ (64,48pad) -> f32 xdbl
        wmma_gemm_kernel<64, 0><<<dim3(MT4, 3), 128, 0, stream>>>(
            xc_bf[dir], WtX + dir * 64 * 48, xdbl[dir], nullptr, 48, 48, 0, 0);
        // selective scan + gate -> yg (bf16)
        scan_kernel<<<(BB * DIN) / 8, 256, 0, stream>>>(
            xdbl[dir], xc_bf[dir], xz_bf[dir], Aneg + dir * DIN * NS,
            m_dtw[dir], m_dtb[dir], m_D[dir], yg_bf[dir]);
        // out_proj: (M,64) @ (64,32) -> f32 ycat columns [dir*32, dir*32+32)
        wmma_gemm_kernel<64, 0><<<dim3(MT4, 2), 128, 0, stream>>>(
            yg_bf[dir], WtOut + dir * 64 * 32, ycat, nullptr, 32, 64, dir * 32, 0);
    }

    final_kernel<<<MTOT / 8, 256, 0, stream>>>(ycat, x, w1n, c1b, idw, idb, w2n, c2b,
                                               (float*)d_out);
}
